// Encoder_28535762715315
// MI455X (gfx1250) — matrix operand order repaired and checked by reading, not compiled, not run on hardware
//
#include <hip/hip_runtime.h>
#include <cstdint>
#include <cstddef>

// ---------------------------------------------------------------------------
// MI455X / gfx1250 encoder forward.
//  - All GEMMs: v_wmma_f32_16x16x32_bf16 (wave32), f32 accumulation.
//  - Weights pre-transposed + pre-converted to bf16 once per launch.
//  - GEMM is double-buffered: B tiles via TDM (tensor_load_to_lds, TENSORcnt,
//    LDS padding via D# pad fields), A tiles via f32->bf16 cvt staging,
//    next tile staged while current tile is computed.
//  - Attention K tiles via GLOBAL_LOAD_ASYNC_TO_LDS_B128 (ASYNCcnt).
//  - Attention is flash-style (no SxS materialization).
// ---------------------------------------------------------------------------

typedef __attribute__((ext_vector_type(4)))  float    vf4;
typedef __attribute__((ext_vector_type(8)))  float    v8f;
typedef __attribute__((ext_vector_type(16))) __bf16   v16bf;
typedef __attribute__((ext_vector_type(4)))  unsigned vu4;
typedef __attribute__((ext_vector_type(8)))  unsigned vu8;

union Frag {
  v16bf  v;
  vf4    q[2];
  __bf16 e[16];
};

#define D_MODEL 512
#define D_FF    2048
#define N_HEADS 8
#define D_HEAD  64
#define SEQ     1024
#define BATCH   4
#define ROWS    (BATCH * SEQ)   // 4096 token rows

// ---------------------- CDNA5 async global->LDS copy -----------------------
// 16B per lane, tracked by ASYNCcnt.  LDS operand = low 32 bits of the flat
// pointer (DS adds LDS_BASE; ISA 10.2 truncates flat->LDS).
__device__ inline void async_copy_b128(const void* lds_dst, const void* gsrc) {
  const unsigned la           = (unsigned)(uintptr_t)lds_dst;
  const unsigned long long ga = (unsigned long long)(uintptr_t)gsrc;
  asm volatile("global_load_async_to_lds_b128 %0, %1, off"
               :: "v"(la), "v"(ga) : "memory");
}
__device__ inline void wait_async0() {
  asm volatile("s_wait_asynccnt 0x0" ::: "memory");
}

// --------------------------- CDNA5 TDM (2D tile) ---------------------------
// One instruction DMAs a row_elems x col_elems bf16 tile (global row stride
// g_stride elems) into LDS, inserting 32B of pad after every 64B row so the
// LDS layout has 48-bf16 row stride.  D# packing per ISA 8.3/8.4.
__device__ inline void tdm_load_2d_bf16(const void* lds_dst, const void* gsrc,
                                        unsigned row_elems /*tile_dim1*/,
                                        unsigned col_elems /*tile_dim0*/,
                                        unsigned g_stride  /*elems*/) {
  const unsigned long long ga = (unsigned long long)(uintptr_t)gsrc;
  vu4 g0;
  g0[0] = 1u;                                   // count=1, user descriptor
  g0[1] = (unsigned)(uintptr_t)lds_dst;         // lds_addr (bytes)
  g0[2] = (unsigned)ga;                         // global_addr[31:0]
  g0[3] = (unsigned)((ga >> 32) & 0x1FFFFFFu)   // global_addr[56:32]
        | (2u << 30);                           // type = 2 ("image")
  vu8 g1;
  g1[0] = (1u << 16)                            // data_size = 2 bytes
        | (1u << 20)                            // pad_enable
        | (3u << 22)                            // pad_interval: 16 DW (64B)
        | (7u << 25);                           // pad_amount:    8 DW (32B)
  g1[1] = (col_elems & 0xFFFFu) << 16;                               // tensor_dim0 lo
  g1[2] = ((col_elems >> 16) & 0xFFFFu) | ((row_elems & 0xFFFFu) << 16); // dim0 hi | dim1 lo
  g1[3] = ((row_elems >> 16) & 0xFFFFu) | ((col_elems & 0xFFFFu) << 16); // dim1 hi | tile_dim0
  g1[4] = row_elems & 0xFFFFu;                  // tile_dim1 (tile_dim2 = 0)
  g1[5] = g_stride;                             // tensor_dim0_stride[31:0]
  g1[6] = 0u;                                   // stride hi | dim1_stride lo
  g1[7] = 0u;
  asm volatile("tensor_load_to_lds %0, %1" :: "s"(g0), "s"(g1) : "memory");
}
__device__ inline void tensor_wait0() {
#if __has_builtin(__builtin_amdgcn_s_wait_tensorcnt)
  __builtin_amdgcn_s_wait_tensorcnt(0);
#else
  asm volatile("s_wait_tensorcnt 0x0" ::: "memory");
#endif
}
__device__ inline void tensor_wait1() {
#if __has_builtin(__builtin_amdgcn_s_wait_tensorcnt)
  __builtin_amdgcn_s_wait_tensorcnt(1);
#else
  asm volatile("s_wait_tensorcnt 0x1" ::: "memory");
#endif
}

// -------------------------------- reductions -------------------------------
// WMMA C/D rows live in 16-lane halves (lanes 0-15: M=e, lanes 16-31: M=e+8),
// so row reductions are xor-shuffles over masks 1..8.
__device__ inline float half16_max(float v) {
  #pragma unroll
  for (int m = 1; m < 16; m <<= 1) v = fmaxf(v, __shfl_xor(v, m, 32));
  return v;
}
__device__ inline float half16_sum(float v) {
  #pragma unroll
  for (int m = 1; m < 16; m <<= 1) v += __shfl_xor(v, m, 32);
  return v;
}
__device__ inline float block_sum(float v, float* red, int tid) {
  #pragma unroll
  for (int m = 1; m < 32; m <<= 1) v += __shfl_xor(v, m, 32);
  const int wave = tid >> 5, lane = tid & 31;
  if (lane == 0) red[wave] = v;
  __syncthreads();
  float s = 0.f;
  #pragma unroll
  for (int i = 0; i < 8; ++i) s += red[i];
  __syncthreads();
  return s;
}

// ----------------- weight transpose + f32->bf16 conversion -----------------
// W[K][N] f32 -> Wt[N][K] bf16, once per launch, so GEMM B-tiles are pure
// 2D copies (TDM-able).  grid = (K/32, N/32).
__global__ __launch_bounds__(256) void transpose_convert_bf16(
    const float* __restrict__ W, __bf16* __restrict__ Wt, int K, int N) {
  __shared__ float tile[32][33];
  const int k0 = blockIdx.x * 32;
  const int n0 = blockIdx.y * 32;
  const int tid = threadIdx.x;
  const int r = tid >> 3;           // 0..31
  const int c = (tid & 7) << 2;     // 0..28
  vf4 f = *(const vf4*)(W + (size_t)(k0 + r) * N + n0 + c);
  tile[r][c + 0] = f[0]; tile[r][c + 1] = f[1];
  tile[r][c + 2] = f[2]; tile[r][c + 3] = f[3];
  __syncthreads();
  __bf16* o = Wt + (size_t)(n0 + r) * K + k0 + c;
  o[0] = (__bf16)tile[c + 0][r]; o[1] = (__bf16)tile[c + 1][r];
  o[2] = (__bf16)tile[c + 2][r]; o[3] = (__bf16)tile[c + 3][r];
}

// ------------------------- embedding + positional --------------------------
__global__ __launch_bounds__(256) void embed_pe_kernel(
    const int* __restrict__ tokens, const float* __restrict__ emb,
    float* __restrict__ x) {
  const int row = blockIdx.x;            // [0, 4096)
  const int s   = row & (SEQ - 1);
  const int tok = tokens[row];
  const int tid = threadIdx.x;
  for (int d = tid; d < D_MODEL; d += 256) {
    const float inv = __expf(-((float)d * (1.f / (float)D_MODEL)) * 9.2103403719761836f);
    const float ang = (float)s * inv;
    const float pe  = (d & 1) ? __cosf(ang) : __sinf(ang);
    x[(size_t)row * D_MODEL + d] = emb[(size_t)tok * D_MODEL + d] + pe;
  }
}

// ------------------------------ WMMA GEMM ----------------------------------
// C[M,N] = act(A[M,K] @ W + bias), W pre-transposed bf16 as Wt[N][K].
// Block tile 128x128, k-step 32, 8 waves in 2x4, wave tile 64x32 (4x2 WMMA).
// Double-buffered: tile kt+1 staged (TDM for B, cvt-stores for A) while tile
// kt is computed; tensorcnt<=1 guarantees the current B tile has landed.
__global__ __launch_bounds__(256) void gemm_bias_act(
    const float* __restrict__ A, const __bf16* __restrict__ Wt,
    const float* __restrict__ bias, float* __restrict__ C,
    __bf16* __restrict__ Cbf, int M, int N, int K, int relu) {
  constexpr int MT = 128, NT = 128, AST = 48, BST = 48;  // 96B rows: 16B aligned
  __shared__ __align__(16) __bf16 As[2][MT * AST];  // A tiles [128 m][32 k]
  __shared__ __align__(16) __bf16 Bt[2][NT * BST];  // W tiles [128 n][32 k]
  (void)M;

  const int tid  = threadIdx.x;
  const int lane = tid & 31;
  const int wave = tid >> 5;
  const int wm = wave >> 2, wn = wave & 3;
  const int hh = lane >> 4, ml = lane & 15;
  const int m0 = blockIdx.y * MT;
  const int n0 = blockIdx.x * NT;

  v8f acc[4][2];
  #pragma unroll
  for (int i = 0; i < 4; ++i)
    #pragma unroll
    for (int j = 0; j < 2; ++j)
      #pragma unroll
      for (int e = 0; e < 8; ++e) acc[i][j][e] = 0.f;

  const int ksteps = K >> 5;

  auto stage_A = [&](int kt, int buf) {
    const int r = tid >> 1;
    const int c = (tid & 1) << 4;
    const float* g = A + (size_t)(m0 + r) * K + (kt << 5) + c;
    vf4 f0 = *(const vf4*)(g + 0),  f1 = *(const vf4*)(g + 4);
    vf4 f2 = *(const vf4*)(g + 8),  f3 = *(const vf4*)(g + 12);
    __bf16* d = &As[buf][r * AST + c];
    #pragma unroll
    for (int j = 0; j < 4; ++j) {
      d[j] = (__bf16)f0[j]; d[4 + j] = (__bf16)f1[j];
      d[8 + j] = (__bf16)f2[j]; d[12 + j] = (__bf16)f3[j];
    }
    if (kt + 1 < ksteps) __builtin_prefetch(g + 32, 0, 1);
  };
  auto issue_B = [&](int kt, int buf) {  // wave 0 only (EXEC ignored by TDM)
    tdm_load_2d_bf16(&Bt[buf][0], Wt + (size_t)n0 * K + (kt << 5),
                     (unsigned)NT, 32u, (unsigned)K);
  };

  // prologue: stage tile 0
  stage_A(0, 0);
  if (wave == 0) issue_B(0, 0);

  for (int kt = 0; kt < ksteps; ++kt) {
    const int cur = kt & 1, nxt = cur ^ 1;
    if (kt + 1 < ksteps) {          // stage next tile while current computes
      stage_A(kt + 1, nxt);
      if (wave == 0) issue_B(kt + 1, nxt);
      tensor_wait1();               // current B tile complete (in-order TDM)
    } else {
      tensor_wait0();
    }
    __syncthreads();                // staging of cur visible to all waves

    Frag af[4], bf[2];
    #pragma unroll
    for (int sm = 0; sm < 4; ++sm) {
      const __bf16* p = &As[cur][(wm * 64 + sm * 16 + ml) * AST + 8 * hh];
      af[sm].q[0] = *(const vf4*)p;
      af[sm].q[1] = *(const vf4*)(p + 16);
    }
    #pragma unroll
    for (int sn = 0; sn < 2; ++sn) {
      const __bf16* p = &Bt[cur][(wn * 32 + sn * 16 + ml) * BST + 8 * hh];
      bf[sn].q[0] = *(const vf4*)p;
      bf[sn].q[1] = *(const vf4*)(p + 16);
    }
    #pragma unroll
    for (int sm = 0; sm < 4; ++sm)
      #pragma unroll
      for (int sn = 0; sn < 2; ++sn)
        acc[sm][sn] = __builtin_amdgcn_wmma_f32_16x16x32_bf16(
            false, af[sm].v, false, bf[sn].v, (short)0, acc[sm][sn], false, false);
    __syncthreads();                // cur reads done before it is restaged
  }

  // epilogue: C/D layout -> element e is row (8*hh + e), col ml
  #pragma unroll
  for (int sm = 0; sm < 4; ++sm) {
    #pragma unroll
    for (int sn = 0; sn < 2; ++sn) {
      const int col = n0 + wn * 32 + sn * 16 + ml;
      const float bv = bias[col];
      #pragma unroll
      for (int e = 0; e < 8; ++e) {
        const int row = m0 + wm * 64 + sm * 16 + 8 * hh + e;
        float vv = acc[sm][sn][e] + bv;
        if (relu) vv = fmaxf(vv, 0.f);
        C[(size_t)row * N + col] = vv;
        if (Cbf) Cbf[(size_t)row * N + col] = (__bf16)vv;
      }
    }
  }
}

// --------------------------- flash attention -------------------------------
// One block = (b, h, 128 q-rows); 8 waves x 16 q-rows; KV tiles of 64.
// K consumed from its bf16 copy via async-to-LDS (row-major == score B-frag
// layout).  V needs a transpose -> manual stage from f32.
// Output in [B][H][S][dh] flat layout (reference reshape bug preserved).
__global__ __launch_bounds__(256) void attention_kernel(
    const float* __restrict__ Q, const __bf16* __restrict__ Kbf,
    const float* __restrict__ Vb, float* __restrict__ O) {
  constexpr int KVT = 64, KST = 72, VST = 72;  // 144B rows: 16B aligned
  __shared__ __align__(16) __bf16 Ks[KVT * KST];        // K tile  [kv][dh]
  __shared__ __align__(16) __bf16 Vt[D_HEAD * VST];     // V tile  [dh][kv]
  __shared__ __align__(16) __bf16 Pb[8 * 16 * D_HEAD];  // per-wave P [16][64]

  const int tid = threadIdx.x, lane = tid & 31, wave = tid >> 5;
  const int hh = lane >> 4, ml = lane & 15;
  const int bh = blockIdx.x, qblk = blockIdx.y;
  const int b = bh >> 3, h = bh & 7;
  const int s0 = qblk * 128;

  // Q fragments (dh=64 -> two k-steps), loaded once from global f32
  const float* qrow = Q + (size_t)(b * SEQ + s0 + wave * 16 + ml) * D_MODEL + h * D_HEAD;
  Frag aq[2];
  #pragma unroll
  for (int ks = 0; ks < 2; ++ks) {
    const float* p0 = qrow + ks * 32 + 8 * hh;
    const float* p1 = p0 + 16;
    vf4 a0 = *(const vf4*)p0, a1 = *(const vf4*)(p0 + 4);
    vf4 b0 = *(const vf4*)p1, b1 = *(const vf4*)(p1 + 4);
    #pragma unroll
    for (int j = 0; j < 4; ++j) {
      aq[ks].e[j]      = (__bf16)a0[j]; aq[ks].e[4 + j]  = (__bf16)a1[j];
      aq[ks].e[8 + j]  = (__bf16)b0[j]; aq[ks].e[12 + j] = (__bf16)b1[j];
    }
  }

  v8f oacc[4];
  #pragma unroll
  for (int t = 0; t < 4; ++t)
    #pragma unroll
    for (int e = 0; e < 8; ++e) oacc[t][e] = 0.f;
  float mrow[8], lrow[8];
  #pragma unroll
  for (int e = 0; e < 8; ++e) { mrow[e] = -3.4e38f; lrow[e] = 0.f; }

  for (int t0 = 0; t0 < SEQ; t0 += KVT) {
    __syncthreads();  // previous tile fully consumed
    {  // K tile: 64x64 bf16, async copy (2x16B/thread)
      #pragma unroll
      for (int u = 0; u < 2; ++u) {
        const int cch = tid * 2 + u;          // 0..511
        const int r   = cch >> 3;             // 0..63
        const int kc  = (cch & 7) << 3;       // 0..56 bf16 elems
        async_copy_b128(&Ks[r * KST + kc],
                        Kbf + (size_t)(b * SEQ + t0 + r) * D_MODEL + h * D_HEAD + kc);
      }
    }
    {  // V tile transposed + converted, 16 f32 per thread
      const int r = tid >> 2;
      const int c = (tid & 3) << 4;
      const float* gv = Vb + (size_t)(b * SEQ + t0 + r) * D_MODEL + h * D_HEAD + c;
      vf4 v0 = *(const vf4*)gv,       v1 = *(const vf4*)(gv + 4);
      vf4 v2 = *(const vf4*)(gv + 8), v3 = *(const vf4*)(gv + 12);
      #pragma unroll
      for (int j = 0; j < 4; ++j) {
        Vt[(c + j     ) * VST + r] = (__bf16)v0[j];
        Vt[(c + 4 + j ) * VST + r] = (__bf16)v1[j];
        Vt[(c + 8 + j ) * VST + r] = (__bf16)v2[j];
        Vt[(c + 12 + j) * VST + r] = (__bf16)v3[j];
      }
    }
    wait_async0();
    __syncthreads();

    // S = Q @ K^T
    v8f sacc[4];
    #pragma unroll
    for (int t = 0; t < 4; ++t)
      #pragma unroll
      for (int e = 0; e < 8; ++e) sacc[t][e] = 0.f;
    #pragma unroll
    for (int ks = 0; ks < 2; ++ks) {
      #pragma unroll
      for (int t = 0; t < 4; ++t) {
        Frag bk;
        const __bf16* p = &Ks[(t * 16 + ml) * KST + ks * 32 + 8 * hh];
        bk.q[0] = *(const vf4*)p;
        bk.q[1] = *(const vf4*)(p + 16);
        sacc[t] = __builtin_amdgcn_wmma_f32_16x16x32_bf16(
            false, aq[ks].v, false, bk.v, (short)0, sacc[t], false, false);
      }
    }

    // online softmax (scale 1/sqrt(64) = 0.125); row r = 8*hh + e
    float fac[8];
    #pragma unroll
    for (int e = 0; e < 8; ++e) {
      float rv = fmaxf(fmaxf(sacc[0][e], sacc[1][e]),
                       fmaxf(sacc[2][e], sacc[3][e])) * 0.125f;
      rv = half16_max(rv);
      const float nm = fmaxf(mrow[e], rv);
      fac[e] = __expf(mrow[e] - nm);
      mrow[e] = nm;
      lrow[e] *= fac[e];
    }
    #pragma unroll
    for (int t = 0; t < 4; ++t)
      #pragma unroll
      for (int e = 0; e < 8; ++e) oacc[t][e] *= fac[e];

    float rs[8];
    #pragma unroll
    for (int e = 0; e < 8; ++e) rs[e] = 0.f;
    #pragma unroll
    for (int t = 0; t < 4; ++t) {
      #pragma unroll
      for (int e = 0; e < 8; ++e) {
        const float pv = __expf(sacc[t][e] * 0.125f - mrow[e]);
        rs[e] += pv;
        Pb[wave * 1024 + (8 * hh + e) * D_HEAD + t * 16 + ml] = (__bf16)pv;
      }
    }
    #pragma unroll
    for (int e = 0; e < 8; ++e) lrow[e] += half16_sum(rs[e]);
    __syncthreads();  // P visible; also gates next K/V restage

    // O += P @ V
    #pragma unroll
    for (int ks = 0; ks < 2; ++ks) {
      Frag ap;
      const __bf16* p = &Pb[wave * 1024 + ml * D_HEAD + ks * 32 + 8 * hh];
      ap.q[0] = *(const vf4*)p;
      ap.q[1] = *(const vf4*)(p + 16);
      #pragma unroll
      for (int t = 0; t < 4; ++t) {
        Frag bv;
        const __bf16* vp = &Vt[(t * 16 + ml) * VST + ks * 32 + 8 * hh];
        bv.q[0] = *(const vf4*)vp;
        bv.q[1] = *(const vf4*)(vp + 16);
        oacc[t] = __builtin_amdgcn_wmma_f32_16x16x32_bf16(
            false, ap.v, false, bv.v, (short)0, oacc[t], false, false);
      }
    }
  }

  #pragma unroll
  for (int t = 0; t < 4; ++t) {
    #pragma unroll
    for (int e = 0; e < 8; ++e) {
      const int s = s0 + wave * 16 + 8 * hh + e;
      const int d = t * 16 + ml;
      O[((size_t)(b * N_HEADS + h) * SEQ + s) * D_HEAD + d] = oacc[t][e] / lrow[e];
    }
  }
}

// --------------------------- residual + layernorm --------------------------
__global__ __launch_bounds__(256) void add_layernorm_kernel(
    const float* __restrict__ delta, const float* __restrict__ g,
    const float* __restrict__ beta, float* __restrict__ x) {
  __shared__ float red[8];
  const int row = blockIdx.x, tid = threadIdx.x;
  float* xr = x + (size_t)row * D_MODEL;
  const float* dr = delta + (size_t)row * D_MODEL;
  const float v0 = xr[tid] + dr[tid];
  const float v1 = xr[tid + 256] + dr[tid + 256];
  const float mean = block_sum(v0 + v1, red, tid) * (1.f / (float)D_MODEL);
  const float d0 = v0 - mean, d1 = v1 - mean;
  const float var = block_sum(d0 * d0 + d1 * d1, red, tid) * (1.f / (float)D_MODEL);
  const float rstd = rsqrtf(var + 1e-5f);
  xr[tid]       = d0 * rstd * g[tid]       + beta[tid];
  xr[tid + 256] = d1 * rstd * g[tid + 256] + beta[tid + 256];
}

// ------------------------------- driver ------------------------------------
extern "C" void kernel_launch(void* const* d_in, const int* in_sizes, int n_in,
                              void* d_out, int out_size, void* d_ws, size_t ws_size,
                              hipStream_t stream) {
  (void)in_sizes; (void)n_in; (void)out_size; (void)ws_size;
  const int*   tokens = (const int*)d_in[0];
  const float* emb    = (const float*)d_in[1];
  const float* Wq = (const float*)d_in[2];
  const float* bq = (const float*)d_in[3];
  const float* Wk = (const float*)d_in[4];
  const float* bk = (const float*)d_in[5];
  const float* Wv = (const float*)d_in[6];
  const float* bv = (const float*)d_in[7];
  const float* Wo = (const float*)d_in[8];
  const float* bo = (const float*)d_in[9];
  const float* W1 = (const float*)d_in[10];
  const float* b1 = (const float*)d_in[11];
  const float* W2 = (const float*)d_in[12];
  const float* b2 = (const float*)d_in[13];
  const float* lng = (const float*)d_in[14];
  const float* lnb = (const float*)d_in[15];

  const size_t RD = (size_t)ROWS * D_MODEL;  // 2M elems
  float* ws   = (float*)d_ws;
  float* x    = ws;            // [4096,512] f32
  float* tmp  = x + RD;
  float* q    = tmp + RD;
  float* k    = q + RD;
  float* v    = k + RD;
  float* ab   = v + RD;        // [B,H,S,dh] flat == [4096,512]
  float* hid  = ab + RD;       // [4096,2048] f32
  __bf16* kbf    = (__bf16*)(hid + (size_t)ROWS * D_FF);  // bf16 copy of K
  __bf16* warena = kbf + RD;                              // bf16 weight arena

  const size_t SQ = (size_t)D_MODEL * D_MODEL;  // 262144
  const size_t FF = (size_t)D_MODEL * D_FF;     // 1048576
  const size_t PER_LAYER = 4 * SQ + 2 * FF;     // bf16 elems per layer

  // ---- one-time per launch: transpose+convert all weights to bf16 ----
  for (int l = 0; l < 6; ++l) {
    __bf16* wqT = warena + (size_t)l * PER_LAYER;
    __bf16* wkT = wqT + SQ;
    __bf16* wvT = wkT + SQ;
    __bf16* woT = wvT + SQ;
    __bf16* w1T = woT + SQ;   // [2048][512]
    __bf16* w2T = w1T + FF;   // [512][2048]
    const dim3 gSq(D_MODEL / 32, D_MODEL / 32);
    transpose_convert_bf16<<<gSq, 256, 0, stream>>>(Wq + l * SQ, wqT, D_MODEL, D_MODEL);
    transpose_convert_bf16<<<gSq, 256, 0, stream>>>(Wk + l * SQ, wkT, D_MODEL, D_MODEL);
    transpose_convert_bf16<<<gSq, 256, 0, stream>>>(Wv + l * SQ, wvT, D_MODEL, D_MODEL);
    transpose_convert_bf16<<<gSq, 256, 0, stream>>>(Wo + l * SQ, woT, D_MODEL, D_MODEL);
    transpose_convert_bf16<<<dim3(D_MODEL / 32, D_FF / 32), 256, 0, stream>>>(
        W1 + l * FF, w1T, D_MODEL, D_FF);
    transpose_convert_bf16<<<dim3(D_FF / 32, D_MODEL / 32), 256, 0, stream>>>(
        W2 + l * FF, w2T, D_FF, D_MODEL);
  }

  embed_pe_kernel<<<ROWS, 256, 0, stream>>>(tokens, emb, x);

  const dim3 g512(D_MODEL / 128, ROWS / 128);
  const dim3 g2048(D_FF / 128, ROWS / 128);
  const dim3 gattn(BATCH * N_HEADS, SEQ / 128);

  for (int l = 0; l < 6; ++l) {
    __bf16* wqT = warena + (size_t)l * PER_LAYER;
    __bf16* wkT = wqT + SQ;
    __bf16* wvT = wkT + SQ;
    __bf16* woT = wvT + SQ;
    __bf16* w1T = woT + SQ;
    __bf16* w2T = w1T + FF;
    const float* gl = lng + (size_t)l * D_MODEL;
    const float* bl = lnb + (size_t)l * D_MODEL;

    gemm_bias_act<<<g512, 256, 0, stream>>>(x, wqT, bq + (size_t)l * D_MODEL, q,
                                            nullptr, ROWS, D_MODEL, D_MODEL, 0);
    gemm_bias_act<<<g512, 256, 0, stream>>>(x, wkT, bk + (size_t)l * D_MODEL, k,
                                            kbf, ROWS, D_MODEL, D_MODEL, 0);
    gemm_bias_act<<<g512, 256, 0, stream>>>(x, wvT, bv + (size_t)l * D_MODEL, v,
                                            nullptr, ROWS, D_MODEL, D_MODEL, 0);
    attention_kernel<<<gattn, 256, 0, stream>>>(q, kbf, v, ab);
    gemm_bias_act<<<g512, 256, 0, stream>>>(ab, woT, bo + (size_t)l * D_MODEL, tmp,
                                            nullptr, ROWS, D_MODEL, D_MODEL, 0);
    add_layernorm_kernel<<<ROWS, 256, 0, stream>>>(tmp, gl, bl, x);
    gemm_bias_act<<<g2048, 256, 0, stream>>>(x, w1T, b1 + (size_t)l * D_FF, hid,
                                             nullptr, ROWS, D_FF, D_MODEL, 1);
    gemm_bias_act<<<g512, 256, 0, stream>>>(hid, w2T, b2 + (size_t)l * D_MODEL, tmp,
                                            nullptr, ROWS, D_MODEL, D_FF, 0);
    add_layernorm_kernel<<<ROWS, 256, 0, stream>>>(tmp, gl, bl, x);
  }

  hipMemcpyAsync(d_out, x, RD * sizeof(float), hipMemcpyDeviceToDevice, stream);
}
